// Locoformer_962072674623
// MI455X (gfx1250) — matrix operand order repaired and checked by reading, not compiled, not run on hardware
//
#include <hip/hip_runtime.h>

typedef __attribute__((ext_vector_type(16))) __bf16 v16bf;
typedef __attribute__((ext_vector_type(8)))  __bf16 v8bf;
typedef __attribute__((ext_vector_type(8)))  float  v8f;
typedef __attribute__((ext_vector_type(4)))  unsigned int tdm_u32x4;
typedef __attribute__((ext_vector_type(4)))  int          tdm_i32x4;
typedef __attribute__((ext_vector_type(8)))  int          tdm_i32x8;

#define HEADS   16
#define DH      64
#define NSEQ    2048
#define BATCH   2
#define DMODEL  1024
#define WINDOW  512
#define MROWS   (BATCH*NSEQ)   // 4096 token rows

#define CDNA5_HAS_TDM (__has_builtin(__builtin_amdgcn_tensor_load_to_lds) && \
                       __has_builtin(__builtin_amdgcn_s_wait_tensorcnt))

// ---------------------------------------------------------------------------
// WMMA helpers for v_wmma_f32_16x16x32_bf16 (CDNA5 / gfx1250, wave32)
//
// A operand (16x32, MxK): lane L holds row m=L&15; half=L>>4;
//   a[i] is K = half*16 + i                      -> one contiguous 16-elem run
// B operand (32x16, KxN): lane L holds col n=L&15;
//   b[i] is K = half*16 + i                      -> one contiguous 16-elem run
// C/D (16x16 f32): lane L col n=L&15; reg r holds row (r + 8*half)
// ---------------------------------------------------------------------------
__device__ __forceinline__ v16bf load_a16(const __bf16* p, int half) {
  return *(const v16bf*)(p + half * 16);
}
__device__ __forceinline__ v16bf load_b16(const __bf16* p, int half) {
  return *(const v16bf*)(p + half * 16);
}
__device__ __forceinline__ v8f wmma_bf16(v16bf a, v16bf b, v8f c) {
  return __builtin_amdgcn_wmma_f32_16x16x32_bf16(false, a, false, b,
                                                 (short)0, c, false, false);
}

// ---------------------------------------------------------------------------
// TDM: async-DMA a 16-row x 1024-col bf16 tile (row stride 1024 elems) from
// global memory into LDS.  2-D descriptor, data_size = 2 bytes.
//   D# group0: count=1 | lds_addr | global_addr(57b) | type=2
//   D# group1: data_size=1(2B), tensor_dim0=1024, tensor_dim1=4096,
//              tile_dim0=1024, tile_dim1=16, tensor_dim0_stride=1024
// ---------------------------------------------------------------------------
#if CDNA5_HAS_TDM
__device__ __forceinline__ void tdm_load_tile16(const __bf16* gsrc, unsigned lds_off) {
  unsigned long long ga = (unsigned long long)(uintptr_t)gsrc;
  tdm_u32x4 g0;
  g0[0] = 1u;                                   // count=1, load, not restore
  g0[1] = lds_off;                              // LDS byte address
  g0[2] = (unsigned)ga;                         // global_addr[31:0]
  g0[3] = ((unsigned)(ga >> 32) & 0x01ffffffu)  // global_addr[56:32]
          | (2u << 30);                         // type=2 ("image")
  tdm_i32x8 g1;
  g1[0] = 1 << 16;                              // data_size=1 -> 2 bytes
  g1[1] = (int)(1024u << 16);                   // tensor_dim0=1024 @bit48
  g1[2] = (int)(4096u << 16);                   // tensor_dim1=4096 @bit80
  g1[3] = (int)(1024u << 16);                   // tile_dim0=1024  @bit112
  g1[4] = 16;                                   // tile_dim1=16 (tile_dim2=0)
  g1[5] = 1024;                                 // tensor_dim0_stride=1024
  g1[6] = 0;
  g1[7] = 0;
  tdm_i32x4 gz = {0, 0, 0, 0};
#if defined(__clang_major__) && (__clang_major__ >= 23)
  tdm_i32x8 gz8 = {0, 0, 0, 0, 0, 0, 0, 0};
  __builtin_amdgcn_tensor_load_to_lds(g0, g1, gz, gz, gz8, 0);
#else
  __builtin_amdgcn_tensor_load_to_lds(g0, g1, gz, gz, 0);
#endif
}
#endif

// Stage a 16x1024 bf16 A-tile into LDS for the whole workgroup.
__device__ __forceinline__ void stage_a_tile(const __bf16* gsrc, __bf16* aTile,
                                             int wave) {
#if CDNA5_HAS_TDM
  if (wave == 0) {
    tdm_load_tile16(gsrc, (unsigned)(uintptr_t)aTile);
    __builtin_amdgcn_s_wait_tensorcnt(0);
  }
#else
  (void)wave;
  const v8bf* s = (const v8bf*)gsrc;
  v8bf* d = (v8bf*)aTile;
  for (int idx = threadIdx.x; idx < 16 * DMODEL / 8; idx += 128) d[idx] = s[idx];
#endif
  __syncthreads();
}

// ---------------------------------------------------------------------------
// Prep kernels
// ---------------------------------------------------------------------------
__global__ void cvt_tokens(const float* __restrict__ t, __bf16* __restrict__ o, int n) {
  int i = blockIdx.x * 256 + threadIdx.x;
  if (i < n) o[i] = (__bf16)t[i];
}

// WTqkv[c][k], c in [0,3072): c<1024 -> w_q col c; else -> w_kv col (c-1024)
__global__ void build_wqkvT(const float* __restrict__ wq, const float* __restrict__ wkv,
                            __bf16* __restrict__ WT) {
  int i = blockIdx.x * 256 + threadIdx.x;   // 3072*1024 threads
  int c = i >> 10, k = i & 1023;
  float v = (c < DMODEL) ? wq[(size_t)k * DMODEL + c]
                         : wkv[(size_t)k * (2 * DMODEL) + (c - DMODEL)];
  WT[i] = (__bf16)v;
}

__global__ void build_woutT(const float* __restrict__ w, __bf16* __restrict__ WT) {
  int i = blockIdx.x * 256 + threadIdx.x;   // 1024*1024 threads
  int c = i >> 10, k = i & 1023;
  WT[i] = (__bf16)w[(size_t)k * DMODEL + c];
}

// gates[tr][h] = sigmoid(sum_k tokens[tr][k] * w_gates[k][h]); one block per row
__global__ __launch_bounds__(256) void gates_kernel(const float* __restrict__ tok,
                                                    const float* __restrict__ wg,
                                                    float* __restrict__ gates) {
  __shared__ float red[256];
  int tr = blockIdx.x;
  int h = threadIdx.x & 15, chunk = threadIdx.x >> 4;   // 16 chunks x 64 elems
  float s = 0.f;
  for (int e = 0; e < 64; ++e) {
    int kk = chunk * 64 + e;
    s += tok[(size_t)tr * DMODEL + kk] * wg[(size_t)kk * HEADS + h];
  }
  red[threadIdx.x] = s;
  __syncthreads();
  for (int st = 8; st > 0; st >>= 1) {
    if (chunk < st) red[chunk * 16 + h] += red[(chunk + st) * 16 + h];
    __syncthreads();
  }
  if (chunk == 0) gates[(size_t)tr * HEADS + h] = 1.f / (1.f + __expf(-red[h]));
}

// ---------------------------------------------------------------------------
// Fused QKV projection + scale + RoPE.
// One workgroup (4 waves) owns one 16-token row tile; A staged via TDM into
// LDS and shared.  Each wave computes 12 col-groups of 64 (kind/head decode).
// ---------------------------------------------------------------------------
__global__ __launch_bounds__(128) void qkv_wmma(const __bf16* __restrict__ tokBF,
                                                const __bf16* __restrict__ WT,
                                                __bf16* __restrict__ qO,
                                                __bf16* __restrict__ kO,
                                                __bf16* __restrict__ vT) {
  __shared__ __attribute__((aligned(128))) __bf16 aTile[16 * DMODEL];  // 32 KB
  int wave = threadIdx.x >> 5, lane = threadIdx.x & 31;
  int half = lane >> 4, n = lane & 15;
  int rowTile = blockIdx.x;                       // 256 row tiles

  stage_a_tile(tokBF + (size_t)rowTile * 16 * DMODEL, aTile, wave);

  const float scale = 0.125f;  // 64^-0.5
  for (int cc = 0; cc < 12; ++cc) {
    int cg = wave * 12 + cc;                      // 0..47
    v8f acc[4] = {};
    const __bf16* arow = aTile + (size_t)n * DMODEL;
    for (int k = 0; k < DMODEL; k += 32) {
      v16bf a = load_a16(arow + k, half);
#pragma unroll
      for (int t = 0; t < 4; ++t) {
        const __bf16* brow = WT + (size_t)(cg * 64 + t * 16 + n) * DMODEL + k;
        acc[t] = wmma_bf16(a, load_b16(brow, half), acc[t]);
      }
    }

    int kind = cg >> 4;          // 0=Q, 1=K, 2=V
    int h = cg & 15;
#pragma unroll
    for (int t = 0; t < 4; ++t) {
      int dh = t * 16 + n;
      // inv_freq = theta^{-(dh&~1)/64};  ln(10000)=9.21034037
      float invf = __expf(-(float)(dh & ~1) * (9.21034037f / 64.f));
#pragma unroll
      for (int r = 0; r < 8; ++r) {
        int row = rowTile * 16 + r + 8 * half;
        int b = row >> 11;        // / NSEQ
        int pos = row & (NSEQ - 1);
        float val = acc[t][r];
        float partner = __shfl_xor(val, 1, 32);   // element at dh^1, same row
        if (kind <= 1) {
          float rh = (dh & 1) ? partner : -partner;  // rotate_half
          float sn, cs;
          __sincosf((float)pos * invf, &sn, &cs);
          val = val * cs + rh * sn;
          if (kind == 0) val *= scale;
        }
        size_t bh = (size_t)(b * HEADS + h);
        if (kind == 0)      qO[(bh * NSEQ + pos) * DH + dh] = (__bf16)val;
        else if (kind == 1) kO[(bh * NSEQ + pos) * DH + dh] = (__bf16)val;
        else                vT[(bh * DH + dh) * NSEQ + pos] = (__bf16)val;
      }
    }
  }
}

// ---------------------------------------------------------------------------
// Sliding-window flash attention. One wave per 16-query tile of one (b,h).
// ---------------------------------------------------------------------------
__global__ __launch_bounds__(128) void attn_wmma(const __bf16* __restrict__ qB,
                                                 const __bf16* __restrict__ kB,
                                                 const __bf16* __restrict__ vT,
                                                 const float* __restrict__ gates,
                                                 __bf16* __restrict__ aOut) {
  __shared__ __attribute__((aligned(32))) __bf16 pS[4][16 * 32];
  int wave = threadIdx.x >> 5, lane = threadIdx.x & 31;
  int half = lane >> 4, n = lane & 15;
  int gid = blockIdx.x * 4 + wave;          // 4096 wave-tiles total
  int qt = gid & 127;                       // 128 query tiles per (b,h)
  int bh = gid >> 7;                        // 0..31
  int b = bh >> 4, h = bh & 15;
  int qs = qt * 16;

  // Q tile stays in registers for the whole key loop (A operand, dh split 0/32)
  const __bf16* qrow = qB + ((size_t)bh * NSEQ + qs + n) * DH;
  v16bf qa0 = load_a16(qrow, half);
  v16bf qa1 = load_a16(qrow + 32, half);

  v8f o[4] = {};
  float mrow[8], lrow[8];
#pragma unroll
  for (int r = 0; r < 8; ++r) { mrow[r] = -3.0e38f; lrow[r] = 0.f; }

  int kb0 = qs - WINDOW;
  if (kb0 < 0) kb0 = 0;
  kb0 &= ~31;
  __bf16* P = pS[wave];

  for (int kb = kb0; kb <= qs + 15; kb += 32) {
    // prefetch next key block (global_prefetch_b8)
    __builtin_prefetch(kB + ((size_t)bh * NSEQ + kb + 32 + n) * DH, 0, 1);

    // S = q . k^T   (two 16-key subtiles, K=64 -> 2 WMMA steps each)
    v8f s0 = {}, s1 = {};
    {
      const __bf16* kr0 = kB + ((size_t)bh * NSEQ + kb + n) * DH;
      s0 = wmma_bf16(qa0, load_b16(kr0, half), s0);
      s0 = wmma_bf16(qa1, load_b16(kr0 + 32, half), s0);
      const __bf16* kr1 = kB + ((size_t)bh * NSEQ + kb + 16 + n) * DH;
      s1 = wmma_bf16(qa0, load_b16(kr1, half), s1);
      s1 = wmma_bf16(qa1, load_b16(kr1 + 32, half), s1);
    }
    // online masked softmax, row reductions across 16-lane halves
#pragma unroll
    for (int r = 0; r < 8; ++r) {
      int i = qs + r + 8 * half;
      int j0 = kb + n, j1 = kb + 16 + n;
      bool v0 = (j0 <= i) && (i - j0 <= WINDOW);
      bool v1 = (j1 <= i) && (i - j1 <= WINDOW);
      float x0 = v0 ? s0[r] : -3.0e38f;
      float x1 = v1 ? s1[r] : -3.0e38f;
      float mx = fmaxf(x0, x1);
#pragma unroll
      for (int off = 1; off < 16; off <<= 1) mx = fmaxf(mx, __shfl_xor(mx, off, 32));
      float mnew = fmaxf(mrow[r], mx);
      float alpha = __expf(mrow[r] - mnew);
      float p0 = v0 ? __expf(s0[r] - mnew) : 0.f;
      float p1 = v1 ? __expf(s1[r] - mnew) : 0.f;
      float ps = p0 + p1;
#pragma unroll
      for (int off = 1; off < 16; off <<= 1) ps += __shfl_xor(ps, off, 32);
      lrow[r] = lrow[r] * alpha + ps;
      mrow[r] = mnew;
#pragma unroll
      for (int d = 0; d < 4; ++d) o[d][r] *= alpha;
      int row = r + 8 * half;
      P[row * 32 + n]      = (__bf16)p0;        // C layout -> LDS [16][32]
      P[row * 32 + 16 + n] = (__bf16)p1;
    }
    asm volatile("s_wait_dscnt 0x0" ::: "memory");  // LDS stores visible to wave
    v16bf pa = load_a16(P + n * 32, half);          // re-read in A layout
    // O += P . V   (V pre-transposed -> contiguous B loads), dh in 4 N-tiles
#pragma unroll
    for (int d = 0; d < 4; ++d) {
      const __bf16* vr = vT + ((size_t)bh * DH + d * 16 + n) * NSEQ + kb;
      o[d] = wmma_bf16(pa, load_b16(vr, half), o[d]);
    }
  }

  // normalize, gate, store to [B*N][H*DH] bf16 (A operand of output GEMM)
#pragma unroll
  for (int r = 0; r < 8; ++r) {
    int i = qs + r + 8 * half;
    float gi = gates[((size_t)b * NSEQ + i) * HEADS + h] / lrow[r];
#pragma unroll
    for (int d = 0; d < 4; ++d) {
      aOut[((size_t)b * NSEQ + i) * DMODEL + h * DH + d * 16 + n] =
          (__bf16)(o[d][r] * gi);
    }
  }
}

// ---------------------------------------------------------------------------
// Output projection: attn_out(bf16) @ w_out -> fp32 d_out.
// Workgroup owns one 16-row tile (A via TDM->LDS); each wave does 4 col-groups.
// ---------------------------------------------------------------------------
__global__ __launch_bounds__(128) void outproj_wmma(const __bf16* __restrict__ aIn,
                                                    const __bf16* __restrict__ WT,
                                                    float* __restrict__ out) {
  __shared__ __attribute__((aligned(128))) __bf16 aTile[16 * DMODEL];  // 32 KB
  int wave = threadIdx.x >> 5, lane = threadIdx.x & 31;
  int half = lane >> 4, n = lane & 15;
  int rowTile = blockIdx.x;                       // 256 row tiles

  stage_a_tile(aIn + (size_t)rowTile * 16 * DMODEL, aTile, wave);

  for (int cc = 0; cc < 4; ++cc) {
    int cg = wave * 4 + cc;                       // 0..15
    v8f acc[4] = {};
    const __bf16* arow = aTile + (size_t)n * DMODEL;
    for (int k = 0; k < DMODEL; k += 32) {
      v16bf a = load_a16(arow + k, half);
#pragma unroll
      for (int t = 0; t < 4; ++t) {
        const __bf16* brow = WT + (size_t)(cg * 64 + t * 16 + n) * DMODEL + k;
        acc[t] = wmma_bf16(a, load_b16(brow, half), acc[t]);
      }
    }
#pragma unroll
    for (int t = 0; t < 4; ++t) {
#pragma unroll
      for (int r = 0; r < 8; ++r) {
        int row = rowTile * 16 + r + 8 * half;
        out[(size_t)row * DMODEL + cg * 64 + t * 16 + n] = acc[t][r];
      }
    }
  }
}

// ---------------------------------------------------------------------------
extern "C" void kernel_launch(void* const* d_in, const int* in_sizes, int n_in,
                              void* d_out, int out_size, void* d_ws, size_t ws_size,
                              hipStream_t stream) {
  (void)in_sizes; (void)n_in; (void)out_size; (void)ws_size;
  const float* tokens  = (const float*)d_in[0];
  const float* w_q     = (const float*)d_in[1];
  const float* w_kv    = (const float*)d_in[2];
  const float* w_out   = (const float*)d_in[3];
  const float* w_gates = (const float*)d_in[4];
  float* out = (float*)d_out;

  char* ws = (char*)d_ws;                 // ~50.6 MB used
  __bf16* tokBF = (__bf16*)(ws + 0);          //  8 MB  [4096][1024]
  __bf16* WTqkv = (__bf16*)(ws + 8388608);    //  6 MB  [3072][1024] (transposed)
  __bf16* WTout = (__bf16*)(ws + 14680064);   //  2 MB  [1024][1024] (transposed)
  float*  gates = (float*) (ws + 16777216);   // .25 MB [4096][16]
  __bf16* qBuf  = (__bf16*)(ws + 17039360);   //  8 MB  [B][H][N][DH]
  __bf16* kBuf  = (__bf16*)(ws + 25427968);   //  8 MB  [B][H][N][DH]
  __bf16* vTBuf = (__bf16*)(ws + 33816576);   //  8 MB  [B][H][DH][N]
  __bf16* aBuf  = (__bf16*)(ws + 42205184);   //  8 MB  [4096][1024]

  cvt_tokens <<<(MROWS * DMODEL) / 256, 256, 0, stream>>>(tokens, tokBF, MROWS * DMODEL);
  build_wqkvT<<<(3 * DMODEL * DMODEL) / 256, 256, 0, stream>>>(w_q, w_kv, WTqkv);
  build_woutT<<<(DMODEL * DMODEL) / 256, 256, 0, stream>>>(w_out, WTout);
  gates_kernel<<<MROWS, 256, 0, stream>>>(tokens, w_gates, gates);

  qkv_wmma    <<<256, 128, 0, stream>>>(tokBF, WTqkv, qBuf, kBuf, vTBuf);
  attn_wmma   <<<(32 * 128) / 4, 128, 0, stream>>>(qBuf, kBuf, vTBuf, gates, aBuf);
  outproj_wmma<<<256, 128, 0, stream>>>(aBuf, WTout, out);
}
